// GeometricWordLM_9912784520037
// MI455X (gfx1250) — matrix operand order repaired and checked by reading, not compiled, not run on hardware
//
#include <hip/hip_runtime.h>
#include <hip/hip_bf16.h>

typedef _Float16 half8  __attribute__((ext_vector_type(8)));
typedef _Float16 half16 __attribute__((ext_vector_type(16)));
typedef float    float8 __attribute__((ext_vector_type(8)));

#define BATCH  2048
#define SEQ    128
#define DIM    128
#define VOCAB  50257
#define MSEED  200          // N_SEEDS
#define KPAD   224          // 7 * 32 (K padded for 16x16x32 WMMA)
#define KT     7            // k-steps of 32
#define NTILES 3144         // ceil(50257/16)=3142, padded to multiple of 4
#define NGROUPS (NTILES/4)  // 786 : each wave does 4 N-tiles (16x64 strip)

// ---------------------------------------------------------------------------
// Kernel 0: convert output_weights[:200] (f32, row-major 200x50257) into f16,
// pre-swizzled for WMMA B in this code's contiguous k order (same as A). For a 32x16 (KxN) f16 B
// fragment: lanes 0-15 hold K=0..15 (halves 0..15), lanes 16-31 hold K=16..31,
// column = lane&15. Each lane's 16 halves stored contiguously (32B) so the
// GEMM reads one coalesced chunk per lane. Chunk index = (ntile*KT+kt)*32+lane.
// ---------------------------------------------------------------------------
__global__ __launch_bounds__(256) void pack_weights(
    const float* __restrict__ W, _Float16* __restrict__ Bp)
{
    int t = blockIdx.x * blockDim.x + threadIdx.x;
    const int total = KT * NTILES * 32;
    if (t >= total) return;
    int lane  = t & 31;
    int chunk = t >> 5;            // = ntile*KT + kt
    int kt    = chunk % KT;
    int ntile = chunk / KT;
    int n     = ntile * 16 + (lane & 15);
    int kbase = kt * 32 + ((lane >> 4) << 4);   // 0..15 or 16..31 within k-step
    half16 v;
#pragma unroll
    for (int h = 0; h < 16; ++h) {
        int k = kbase + h;
        float x = (k < MSEED && n < VOCAB) ? W[(size_t)k * VOCAB + n] : 0.0f;
        v[h] = (_Float16)x;
    }
    *(half16*)(Bp + (size_t)t * 16) = v;
}

// ---------------------------------------------------------------------------
// Kernel 1: per batch row: embedding gather + pos add (staged in LDS with
// stride 129 to avoid bank conflicts), single-query softmax attention,
// context vector, then RBF activations emitted as f16 A-matrix row (K=224,
// zero-padded past 200). Softmax normalization is folded into the context sum.
// ---------------------------------------------------------------------------
__global__ __launch_bounds__(128) void context_acts(
    const int*   __restrict__ token_ids,
    const float* __restrict__ tok_embed,
    const float* __restrict__ pos_embed,
    const float* __restrict__ query,
    const float* __restrict__ positions,
    _Float16*    __restrict__ Ap)
{
    __shared__ float emb[SEQ][DIM + 1];   // +1 pad: conflict-free column reads
    __shared__ float qsh[DIM];
    __shared__ float red[SEQ];
    __shared__ float wts[SEQ];
    __shared__ float ctx[DIM];
    __shared__ int   toks[SEQ];

    const int tid = threadIdx.x;
    const int b   = blockIdx.x;

    toks[tid] = token_ids[b * SEQ + tid];
    qsh[tid]  = query[tid];
    __syncthreads();

    // gather embeddings: coalesced 512B row loads
    for (int s = 0; s < SEQ; ++s)
        emb[s][tid] = tok_embed[(size_t)toks[s] * DIM + tid] + pos_embed[s * DIM + tid];
    __syncthreads();

    // attention score for s = tid
    float sc = 0.0f;
    for (int d = 0; d < DIM; ++d) sc += emb[tid][d] * qsh[d];
    sc *= 0.08838834764831845f;           // 1/sqrt(128)
    red[tid] = sc;
    __syncthreads();
    for (int off = 64; off > 0; off >>= 1) {
        if (tid < off) red[tid] = fmaxf(red[tid], red[tid + off]);
        __syncthreads();
    }
    float mx = red[0];
    __syncthreads();

    float ex = __expf(sc - mx);
    wts[tid] = ex;
    red[tid] = ex;
    __syncthreads();
    for (int off = 64; off > 0; off >>= 1) {
        if (tid < off) red[tid] += red[tid + off];
        __syncthreads();
    }
    float inv = 1.0f / red[0];
    __syncthreads();

    // context component d = tid (column reads are conflict-free via pad)
    float c = 0.0f;
    for (int s = 0; s < SEQ; ++s) c += wts[s] * emb[s][tid];
    c *= inv;
    ctx[tid] = c;
    red[tid] = c * c;
    __syncthreads();
    for (int off = 64; off > 0; off >>= 1) {
        if (tid < off) red[tid] += red[tid + off];
        __syncthreads();
    }
    float cc = red[0];
    __syncthreads();

    // RBF activations -> f16 A row (zero-padded K tail)
    for (int m = tid; m < KPAD; m += SEQ) {
        float a = 0.0f;
        if (m < MSEED) {
            float dot = 0.0f, pp = 0.0f;
            for (int d = 0; d < DIM; ++d) {
                float p = positions[(size_t)m * DIM + d];
                dot += ctx[d] * p;
                pp  += p * p;
            }
            float d2 = cc - 2.0f * dot + pp;
            a = __expf(-d2 * 0.78125f);   // 1/(2*0.8^2) = 0.78125
        }
        Ap[(size_t)b * KPAD + m] = (_Float16)a;
    }
}

// ---------------------------------------------------------------------------
// Kernel 2: logits = A(2048x224 f16) x Bpacked(224x50304 f16) -> f32.
// Each wave computes a 16x64 strip: 4 accumulators, 7 k-steps, 4 WMMAs/step.
// A fragment per lane: row = lane&15, halves 0..7 = K kbase..kbase+7,
// halves 8..15 = K kbase+8..+15 (kbase = 0 for lanes<16, 16 otherwise)
// -> two 16B row-major loads. B fragment: one contiguous 32B packed chunk.
// ---------------------------------------------------------------------------
__global__ __launch_bounds__(256) void rbf_gemm(
    const _Float16* __restrict__ Ap,
    const _Float16* __restrict__ Bp,
    float* __restrict__ out)
{
    const int lane   = threadIdx.x & 31;
    const int wave   = threadIdx.x >> 5;
    const int mtile  = blockIdx.y;
    const int ngroup = blockIdx.x * 8 + wave;   // wave-uniform -> EXEC stays full
    if (ngroup >= NGROUPS) return;

    const int row   = lane & 15;
    const int kbase = (lane < 16) ? 0 : 16;
    const _Float16* arow = Ap + (size_t)(mtile * 16 + row) * KPAD + kbase;
    const int nt0 = ngroup * 4;

    float8 acc[4] = {{0}, {0}, {0}, {0}};

    for (int kt = 0; kt < KT; ++kt) {
        union { half16 v; half8 h[2]; } a;
        a.h[0] = *(const half8*)(arow + kt * 32);
        a.h[1] = *(const half8*)(arow + kt * 32 + 8);

        if (kt + 1 < KT)  // hint next k-step of B (global_prefetch_b8)
            __builtin_prefetch(Bp + ((size_t)(nt0 * KT + kt + 1) * 32 + lane) * 16, 0, 1);

#pragma unroll
        for (int j = 0; j < 4; ++j) {
            half16 bfrag = *(const half16*)(
                Bp + ((size_t)((nt0 + j) * KT + kt) * 32 + lane) * 16);
            acc[j] = __builtin_amdgcn_wmma_f32_16x16x32_f16(
                false, a.v, false, bfrag, (short)0, acc[j], false, false);
        }
    }

    // D layout: VGPR r -> M = r (lanes 0-15) / r+8 (lanes 16-31), N = lane&15
    const int mrow = mtile * 16 + ((lane < 16) ? 0 : 8);
#pragma unroll
    for (int j = 0; j < 4; ++j) {
        int n = (nt0 + j) * 16 + (lane & 15);
        if (n < VOCAB) {
#pragma unroll
            for (int r = 0; r < 8; ++r)
                out[(size_t)(mrow + r) * VOCAB + n] = acc[j][r];
        }
    }
}

// ---------------------------------------------------------------------------
extern "C" void kernel_launch(void* const* d_in, const int* in_sizes, int n_in,
                              void* d_out, int out_size, void* d_ws, size_t ws_size,
                              hipStream_t stream)
{
    const int*   token_ids = (const int*)  d_in[0];   // (2048,128)
    const float* tok_embed = (const float*)d_in[1];   // (50257,128)
    const float* pos_embed = (const float*)d_in[2];   // (128,128)
    const float* query     = (const float*)d_in[3];   // (128,)
    const float* positions = (const float*)d_in[4];   // (2000,128) -> first 200
    const float* weights   = (const float*)d_in[5];   // (2000,50257) -> first 200
    float*       logits    = (float*)d_out;           // (2048,50257)

    // workspace layout
    const size_t bp_bytes = (size_t)KT * NTILES * 32 * 16 * sizeof(_Float16); // ~22.5 MB
    _Float16* Bp = (_Float16*)d_ws;
    _Float16* Ap = (_Float16*)((char*)d_ws + bp_bytes);                       // 2048*224*2

    // 1) pack + convert weights into WMMA B layout
    {
        int total  = KT * NTILES * 32;
        int blocks = (total + 255) / 256;
        pack_weights<<<blocks, 256, 0, stream>>>(weights, Bp);
    }
    // 2) attention context + RBF activations -> f16 A matrix
    context_acts<<<BATCH, 128, 0, stream>>>(token_ids, tok_embed, pos_embed,
                                            query, positions, Ap);
    // 3) WMMA GEMM -> logits
    {
        dim3 grid((NGROUPS + 7) / 8, BATCH / 16);
        rbf_gemm<<<grid, 256, 0, stream>>>(Ap, Bp, logits);
    }
}
